// MixtureOfExperts_35021163331990
// MI455X (gfx1250) — matrix operand order repaired and checked by reading, not compiled, not run on hardware
//
#include <hip/hip_runtime.h>
#include <hip/hip_bf16.h>

// ---------------------------------------------------------------------------
// MoE dense: out[n,f] = sum_e softmax(x@Wg+bg)[n,e] * ( relu(x@W1[e]+b1[e]) @ W2[e] + b2[e] )[n,f]
// TOKENS=8192, D_IN=1024, E=8, H=2048, D_OUT=1024.
// Compute path: bf16 WMMA (v_wmma_f32_16x16x32_bf16), f32 accumulation,
// double-buffered async global->LDS staging (GLOBAL_LOAD_ASYNC_TO_LDS_B128).
// ---------------------------------------------------------------------------

#define TOKENS 8192
#define D_IN   1024
#define NEXP   8
#define D_OUT  1024
#define HDIM   2048

typedef __bf16 bf16_t;
typedef __attribute__((ext_vector_type(16))) __bf16 v16bf;
typedef __attribute__((ext_vector_type(8)))  __bf16 v8bf;
typedef __attribute__((ext_vector_type(8)))  float  v8f;
typedef __attribute__((ext_vector_type(4)))  float  v4f;

static __device__ __forceinline__ bf16_t to_bf16(float f) {
    unsigned int u = __builtin_bit_cast(unsigned int, f);
    unsigned int r = u + 0x7FFFu + ((u >> 16) & 1u);
    unsigned short h = (unsigned short)(r >> 16);
    return __builtin_bit_cast(bf16_t, h);
}

// ---------------------------------------------------------------------------
// Gating: gates[n][e] = softmax_e( x[n,:] @ Wg[:,e] + bg[e] ).  Wave per token.
// ---------------------------------------------------------------------------
__global__ __launch_bounds__(256) void moe_gating_kernel(
    const float* __restrict__ x, const float* __restrict__ Wg,
    const float* __restrict__ bg, float* __restrict__ gates) {
    const int lane  = threadIdx.x & 31;
    const int wid   = threadIdx.x >> 5;
    const int token = blockIdx.x * 8 + wid;

    float acc[8];
#pragma unroll
    for (int e = 0; e < 8; ++e) acc[e] = 0.0f;

    const float* xrow = x + (size_t)token * D_IN;
#pragma unroll 4
    for (int t = 0; t < D_IN / 32; ++t) {
        int d = t * 32 + lane;
        float xv = xrow[d];
        const v4f* w4 = reinterpret_cast<const v4f*>(Wg + (size_t)d * 8);
        v4f w01 = w4[0], w23 = w4[1];
        acc[0] += xv * w01.x; acc[1] += xv * w01.y;
        acc[2] += xv * w01.z; acc[3] += xv * w01.w;
        acc[4] += xv * w23.x; acc[5] += xv * w23.y;
        acc[6] += xv * w23.z; acc[7] += xv * w23.w;
    }
#pragma unroll
    for (int off = 16; off >= 1; off >>= 1) {
#pragma unroll
        for (int e = 0; e < 8; ++e) acc[e] += __shfl_xor(acc[e], off, 32);
    }
    if (lane == 0) {
        float m = -3.4e38f;
#pragma unroll
        for (int e = 0; e < 8; ++e) { acc[e] += bg[e]; m = fmaxf(m, acc[e]); }
        float s = 0.0f;
#pragma unroll
        for (int e = 0; e < 8; ++e) { acc[e] = __expf(acc[e] - m); s += acc[e]; }
        float inv = 1.0f / s;
#pragma unroll
        for (int e = 0; e < 8; ++e) gates[(size_t)token * 8 + e] = acc[e] * inv;
    }
}

// ---------------------------------------------------------------------------
// f32 -> bf16 elementwise convert (x).
// ---------------------------------------------------------------------------
__global__ __launch_bounds__(256) void convert_f32_bf16_kernel(
    const float* __restrict__ src, bf16_t* __restrict__ dst, int n) {
    int i = (blockIdx.x * 256 + threadIdx.x) * 8;
    if (i + 7 < n) {
        const v4f* s4 = reinterpret_cast<const v4f*>(src + i);
        v4f a = s4[0], b = s4[1];
        bf16_t o[8] = { to_bf16(a.x), to_bf16(a.y), to_bf16(a.z), to_bf16(a.w),
                        to_bf16(b.x), to_bf16(b.y), to_bf16(b.z), to_bf16(b.w) };
#pragma unroll
        for (int k = 0; k < 8; ++k) dst[i + k] = o[k];
    }
}

// ---------------------------------------------------------------------------
// Per-expert tiled transpose + convert: dst[e][c][r] = bf16(src[e][r][c])
// ---------------------------------------------------------------------------
__global__ __launch_bounds__(256) void transpose_bf16_kernel(
    const float* __restrict__ src, bf16_t* __restrict__ dst, int rows, int cols) {
    __shared__ float tile[32][33];
    const size_t eoff = (size_t)blockIdx.z * rows * cols;
    const float* s = src + eoff;
    bf16_t*      d = dst + eoff;

    int x  = blockIdx.x * 32 + threadIdx.x;
    int y0 = blockIdx.y * 32 + threadIdx.y;
#pragma unroll
    for (int i = 0; i < 4; ++i)
        tile[threadIdx.y + 8 * i][threadIdx.x] = s[(size_t)(y0 + 8 * i) * cols + x];
    __syncthreads();
    int xo  = blockIdx.y * 32 + threadIdx.x;
    int yo0 = blockIdx.x * 32 + threadIdx.y;
#pragma unroll
    for (int i = 0; i < 4; ++i)
        d[(size_t)(yo0 + 8 * i) * rows + xo] = to_bf16(tile[threadIdx.x][threadIdx.y + 8 * i]);
}

// ---------------------------------------------------------------------------
// WMMA GEMM core, async double-buffered staging.
// C-tile 128x128 per 256-thread block (8 waves 2(M)x4(N); wave = 64x32 = 4x2 frags).
// A:[M][K] bf16 row-major (lda==K).  Bt:[N][K] bf16 row-major (ldb==K).
// ---------------------------------------------------------------------------
#define BM 128
#define BN 128
#define BK 64

static __device__ __forceinline__ void async_b128(unsigned lds_addr, unsigned goff,
                                                  const bf16_t* gbase) {
    // dsaddr = lds_addr (per-lane byte address); mem = SADDR(base) + VADDR(goff)
    asm volatile("global_load_async_to_lds_b128 %0, %1, %2"
                 :
                 : "v"(lds_addr), "v"(goff),
                   "s"((unsigned long long)(size_t)gbase)
                 : "memory");
}
static __device__ __forceinline__ void wait_async_8() {
    asm volatile("s_wait_asynccnt 0x8" ::: "memory");
}
static __device__ __forceinline__ void wait_async_0() {
    asm volatile("s_wait_asynccnt 0x0" ::: "memory");
}

static __device__ __forceinline__ v16bf frag_a(const bf16_t* As, int row, int kk, int lane) {
    int k0 = (lane >= 16) ? 8 : 0;
    const v8bf* p0 = reinterpret_cast<const v8bf*>(&As[row * BK + kk + k0]);
    const v8bf* p1 = reinterpret_cast<const v8bf*>(&As[row * BK + kk + k0 + 16]);
    v8bf lo = *p0, hi = *p1;
    return __builtin_shufflevector(lo, hi, 0, 1, 2, 3, 4, 5, 6, 7, 8, 9, 10, 11, 12, 13, 14, 15);
}
static __device__ __forceinline__ v16bf frag_b(const bf16_t* Bs, int nrow, int kk, int lane) {
    int kb = (lane >= 16) ? 8 : 0;
    const v8bf* p0 = reinterpret_cast<const v8bf*>(&Bs[nrow * BK + kk + kb]);
    const v8bf* p1 = reinterpret_cast<const v8bf*>(&Bs[nrow * BK + kk + kb + 16]);
    v8bf lo = *p0, hi = *p1;
    return __builtin_shufflevector(lo, hi, 0, 1, 2, 3, 4, 5, 6, 7, 8, 9, 10, 11, 12, 13, 14, 15);
}

static __device__ __forceinline__ void gemm_core(
    const bf16_t* __restrict__ A, const bf16_t* __restrict__ Bt, int K,
    bf16_t (*As)[BM * BK], bf16_t (*Bs)[BN * BK], v8f acc[4][2]) {
    const int tid  = threadIdx.x;
    const int lane = tid & 31;
    const int wid  = tid >> 5;
    const int wm   = (wid & 1) * 64;
    const int wn   = (wid >> 1) * 32;

    // per-thread staging chunk coordinates (4 chunks of 16B for A, 4 for B)
    const bf16_t* Ab = A + (size_t)blockIdx.y * BM * K;   // block-row base
    const bf16_t* Bb = Bt + (size_t)blockIdx.x * BN * K;  // block-col base
    const unsigned asb = (unsigned)(size_t)(void*)&As[0][0];
    const unsigned bsb = (unsigned)(size_t)(void*)&Bs[0][0];

    const int S = K / BK;

    // prologue: issue stage 0 into buffer 0 (8 async ops per wave)
    {
#pragma unroll
        for (int i = 0; i < 4; ++i) {
            int idx = tid + i * 256;
            int r = idx >> 3;
            int c = (idx & 7) * 8;
            unsigned loff = (unsigned)((r * BK + c) * 2);
            unsigned goff = (unsigned)((r * K + c) * 2);
            async_b128(asb + loff, goff, Ab);
            async_b128(bsb + loff, goff, Bb);
        }
    }

    for (int s = 0; s < S; ++s) {
        const int buf = s & 1;
        __syncthreads();                    // compute(s-1) done everywhere -> buf^1 free
        if (s + 1 < S) {
            const int k0n = (s + 1) * BK;
            const unsigned lbuf = (unsigned)((buf ^ 1) * BM * BK * 2);
#pragma unroll
            for (int i = 0; i < 4; ++i) {
                int idx = tid + i * 256;
                int r = idx >> 3;
                int c = (idx & 7) * 8;
                unsigned loff = (unsigned)((r * BK + c) * 2) + lbuf;
                unsigned goff = (unsigned)((r * K + k0n + c) * 2);
                async_b128(asb + loff, goff, Ab);
                async_b128(bsb + loff, goff, Bb);
            }
            wait_async_8();                 // oldest 8 (stage s) have landed
        } else {
            wait_async_0();
        }
        __syncthreads();                    // stage s visible to all waves

        const bf16_t* as = As[buf];
        const bf16_t* bs = Bs[buf];
#pragma unroll
        for (int kk = 0; kk < BK; kk += 32) {
            v16bf af[4], bfr[2];
#pragma unroll
            for (int mf = 0; mf < 4; ++mf) af[mf] = frag_a(as, wm + mf * 16 + (lane & 15), kk, lane);
#pragma unroll
            for (int nf = 0; nf < 2; ++nf) bfr[nf] = frag_b(bs, wn + nf * 16 + (lane & 15), kk, lane);
#pragma unroll
            for (int mf = 0; mf < 4; ++mf)
#pragma unroll
                for (int nf = 0; nf < 2; ++nf)
                    acc[mf][nf] = __builtin_amdgcn_wmma_f32_16x16x32_bf16(
                        false, af[mf], false, bfr[nf], (short)0, acc[mf][nf], false, false);
        }
    }
}

// GEMM1: h = relu(x @ W1[e] + b1[e]) -> bf16   (M=TOKENS, N=HDIM, K=D_IN)
__global__ __launch_bounds__(256) void moe_gemm1_kernel(
    const bf16_t* __restrict__ xb, const bf16_t* __restrict__ w1t_e,
    const float* __restrict__ b1_e, bf16_t* __restrict__ hb) {
    __shared__ __align__(16) bf16_t As[2][BM * BK];
    __shared__ __align__(16) bf16_t Bs[2][BN * BK];
    v8f acc[4][2];
    const v8f z = {0.f, 0.f, 0.f, 0.f, 0.f, 0.f, 0.f, 0.f};
#pragma unroll
    for (int mf = 0; mf < 4; ++mf)
#pragma unroll
        for (int nf = 0; nf < 2; ++nf) acc[mf][nf] = z;

    gemm_core(xb, w1t_e, D_IN, As, Bs, acc);

    const int lane = threadIdx.x & 31;
    const int wid  = threadIdx.x >> 5;
    const int wm = (wid & 1) * 64, wn = (wid >> 1) * 32;
#pragma unroll
    for (int nf = 0; nf < 2; ++nf) {
        int col = blockIdx.x * BN + wn + nf * 16 + (lane & 15);
        float b1v = b1_e[col];
#pragma unroll
        for (int mf = 0; mf < 4; ++mf) {
            int row = blockIdx.y * BM + wm + mf * 16 + ((lane >= 16) ? 8 : 0);
#pragma unroll
            for (int i = 0; i < 8; ++i) {
                float v = acc[mf][nf][i] + b1v;
                v = v > 0.0f ? v : 0.0f;
                hb[(size_t)(row + i) * HDIM + col] = to_bf16(v);
            }
        }
    }
}

// GEMM2: out (+)= gates[:,e] * (h @ W2[e] + b2[e])   (M=TOKENS, N=D_OUT, K=HDIM)
__global__ __launch_bounds__(256) void moe_gemm2_kernel(
    const bf16_t* __restrict__ hb, const bf16_t* __restrict__ w2t_e,
    const float* __restrict__ b2_e, const float* __restrict__ gates,
    float* __restrict__ out, int e, int accumulate) {
    __shared__ __align__(16) bf16_t As[2][BM * BK];
    __shared__ __align__(16) bf16_t Bs[2][BN * BK];
    v8f acc[4][2];
    const v8f z = {0.f, 0.f, 0.f, 0.f, 0.f, 0.f, 0.f, 0.f};
#pragma unroll
    for (int mf = 0; mf < 4; ++mf)
#pragma unroll
        for (int nf = 0; nf < 2; ++nf) acc[mf][nf] = z;

    gemm_core(hb, w2t_e, HDIM, As, Bs, acc);

    const int lane = threadIdx.x & 31;
    const int wid  = threadIdx.x >> 5;
    const int wm = (wid & 1) * 64, wn = (wid >> 1) * 32;
#pragma unroll
    for (int nf = 0; nf < 2; ++nf) {
        int col = blockIdx.x * BN + wn + nf * 16 + (lane & 15);
        float b2v = b2_e[col];
#pragma unroll
        for (int mf = 0; mf < 4; ++mf) {
            int row = blockIdx.y * BM + wm + mf * 16 + ((lane >= 16) ? 8 : 0);
#pragma unroll
            for (int i = 0; i < 8; ++i) {
                float g = gates[(size_t)(row + i) * 8 + e];
                float v = g * (acc[mf][nf][i] + b2v);
                size_t o = (size_t)(row + i) * D_OUT + col;
                if (accumulate) out[o] += v; else out[o] = v;
            }
        }
    }
}

// ---------------------------------------------------------------------------
// Launch
// ---------------------------------------------------------------------------
extern "C" void kernel_launch(void* const* d_in, const int* in_sizes, int n_in,
                              void* d_out, int out_size, void* d_ws, size_t ws_size,
                              hipStream_t stream) {
    const float* x  = (const float*)d_in[0];
    const float* W1 = (const float*)d_in[1];
    const float* b1 = (const float*)d_in[2];
    const float* W2 = (const float*)d_in[3];
    const float* b2 = (const float*)d_in[4];
    const float* Wg = (const float*)d_in[5];
    const float* bg = (const float*)d_in[6];
    float* out = (float*)d_out;

    char* ws = (char*)d_ws;
    size_t off = 0;
    float*  gates = (float*)(ws + off);  off += (size_t)TOKENS * NEXP * 4;
    bf16_t* xb    = (bf16_t*)(ws + off); off += (size_t)TOKENS * D_IN * 2;
    bf16_t* w1t   = (bf16_t*)(ws + off); off += (size_t)NEXP * HDIM * D_IN * 2;
    bf16_t* w2t   = (bf16_t*)(ws + off); off += (size_t)NEXP * D_OUT * HDIM * 2;
    bf16_t* hb    = (bf16_t*)(ws + off); off += (size_t)TOKENS * HDIM * 2;
    if (ws_size < off) return;

    moe_gating_kernel<<<TOKENS / 8, 256, 0, stream>>>(x, Wg, bg, gates);

    {
        int n = TOKENS * D_IN;
        convert_f32_bf16_kernel<<<n / (256 * 8), 256, 0, stream>>>(x, xb, n);
    }
    transpose_bf16_kernel<<<dim3(HDIM / 32, D_IN / 32, NEXP), dim3(32, 8), 0, stream>>>(
        W1, w1t, D_IN, HDIM);     // W1[e][d][h] -> w1t[e][h][d]
    transpose_bf16_kernel<<<dim3(D_OUT / 32, HDIM / 32, NEXP), dim3(32, 8), 0, stream>>>(
        W2, w2t, HDIM, D_OUT);    // W2[e][h][f] -> w2t[e][f][h]

    for (int e = 0; e < NEXP; ++e) {
        const bf16_t* w1e = w1t + (size_t)e * HDIM * D_IN;
        const bf16_t* w2e = w2t + (size_t)e * D_OUT * HDIM;
        moe_gemm1_kernel<<<dim3(HDIM / BN, TOKENS / BM), 256, 0, stream>>>(
            xb, w1e, b1 + (size_t)e * HDIM, hb);
        moe_gemm2_kernel<<<dim3(D_OUT / BN, TOKENS / BM), 256, 0, stream>>>(
            hb, w2e, b2 + (size_t)e * D_OUT, gates, out, e, e != 0);
    }
}
